// MultiHeadCausalAttention_69243462746441
// MI455X (gfx1250) — matrix operand order repaired and checked by reading, not compiled, not run on hardware
//
#include <hip/hip_runtime.h>
#include <cstdint>
#include <cstddef>

// ---------------------------------------------------------------------------
// MultiHeadCausalAttention for MI455X (gfx1250), bf16 WMMA pipeline with
// TDM (tensor_load_to_lds) double-buffered GEMM staging.
// B=4, T=2048, C=1024, H=16, Dh=64.
// ---------------------------------------------------------------------------

typedef __attribute__((ext_vector_type(16))) __bf16 v16bf;
typedef __attribute__((ext_vector_type(8)))  __bf16 v8bf;
typedef __attribute__((ext_vector_type(4)))  __bf16 v4bf;
typedef __attribute__((ext_vector_type(8)))  float  v8f;
typedef __attribute__((ext_vector_type(4)))  unsigned int v4u;
typedef __attribute__((ext_vector_type(8)))  int    v8i;
typedef __attribute__((ext_vector_type(4)))  int    v4i;

#define BATCH   4
#define SEQ     2048
#define DMODEL  1024
#define NHEADS  16
#define HDIM    64
#define BH      (BATCH * NHEADS)       // 64
#define MTOK    (BATCH * SEQ)          // 8192

#if defined(__gfx1250__) && __has_builtin(__builtin_amdgcn_tensor_load_to_lds)
#define USE_TDM 1
#else
#define USE_TDM 0
#endif

static __device__ __forceinline__ v8f wmma_bf16(v16bf a, v16bf b, v8f c) {
  // D = A(16x32 bf16) x B(32x16 bf16) + C(16x16 f32)
  return __builtin_amdgcn_wmma_f32_16x16x32_bf16(false, a, false, b,
                                                 (short)0, c, false, false);
}

#if USE_TDM
// ---------------------------------------------------------------------------
// Issue a 2D TDM load: tile_d1 rows x tile_d0 bf16 elements, row stride
// stride_d0 (elements) in global memory, packed contiguously into LDS at
// byte offset lds_byte_off. Descriptor per CDNA5 ISA ch.8 (D# groups 0/1).
// Must be executed from a scalar-branch region (TDM ignores EXEC).
// ---------------------------------------------------------------------------
static __device__ __forceinline__ void tdm_load_2d(const void* gptr,
                                                   unsigned lds_byte_off,
                                                   unsigned tile_d0,
                                                   unsigned tile_d1,
                                                   unsigned stride_d0) {
  unsigned long long ga = (unsigned long long)(size_t)gptr;
  v4u g0;
  g0[0] = 1u;                                        // count=1, user mode
  g0[1] = lds_byte_off;                              // lds_addr
  g0[2] = (unsigned)(ga & 0xFFFFFFFFu);              // global_addr[31:0]
  g0[3] = (unsigned)((ga >> 32) & 0x1FFFFFFu)        // global_addr[56:32]
        | (2u << 30);                                // type = 2 ("image")
  v8i g1;
  g1[0] = (int)(1u << 16);                           // data_size=1 (2 bytes)
  g1[1] = (int)((tile_d0 & 0xFFFFu) << 16);          // tensor_dim0 lo16
  g1[2] = (int)(((tile_d0 >> 16) & 0xFFFFu)          // tensor_dim0 hi16
        | ((tile_d1 & 0xFFFFu) << 16));              // tensor_dim1 lo16
  g1[3] = (int)(((tile_d1 >> 16) & 0xFFFFu)          // tensor_dim1 hi16
        | ((tile_d0 & 0xFFFFu) << 16));              // tile_dim0
  g1[4] = (int)(tile_d1 & 0xFFFFu);                  // tile_dim1 (tile_dim2=0)
  g1[5] = (int)stride_d0;                            // tensor_dim0_stride lo32
  g1[6] = 0;                                         // stride0 hi16 / stride1 lo16
  g1[7] = 0;                                         // stride1 hi32
  v4i g2 = {0, 0, 0, 0};
  v4i g3 = {0, 0, 0, 0};
#if __clang_major__ >= 23
  v8i g4 = {0, 0, 0, 0, 0, 0, 0, 0};
  __builtin_amdgcn_tensor_load_to_lds(g0, g1, g2, g3, g4, 0);
#else
  __builtin_amdgcn_tensor_load_to_lds(g0, g1, g2, g3, 0);
#endif
}
#endif // USE_TDM

// ---------------------------------------------------------------------------
// f32 -> bf16 conversion (vectorized x4)
// ---------------------------------------------------------------------------
__global__ __launch_bounds__(256) void cvt_f32_bf16(const float* __restrict__ s,
                                                    __bf16* __restrict__ d, int n) {
  int i = (blockIdx.x * 256 + threadIdx.x) * 4;
  if (i + 3 < n) {
    float4 f = *(const float4*)(s + i);
    v4bf o;
    o[0] = (__bf16)f.x; o[1] = (__bf16)f.y; o[2] = (__bf16)f.z; o[3] = (__bf16)f.w;
    *(v4bf*)(d + i) = o;
  }
}

// ---------------------------------------------------------------------------
// Fragment helpers (LDS slab layout: row-major tiles, 32 bf16 per row).
// ---------------------------------------------------------------------------
static __device__ __forceinline__ v16bf load_a_frag_lds(const __bf16* smem,
                                                        int elem_base, int row,
                                                        int half) {
  const __bf16* p = smem + elem_base + row * 32 + half * 16;
  v8bf lo = *(const v8bf*)p;
  v8bf hi = *(const v8bf*)(p + 8);
  v16bf r;
  #pragma unroll
  for (int e = 0; e < 8; e++) { r[e] = lo[e]; r[8 + e] = hi[e]; }
  return r;
}

static __device__ __forceinline__ v16bf load_b_frag_lds(const __bf16* smem,
                                                        int elem_base, int n,
                                                        int half) {
  return *(const v16bf*)(smem + elem_base + n * 32 + half * 16);
}

// ---------------------------------------------------------------------------
// GEMM: C(M,N) = A(M,K) @ W(N,K)^T, bf16 inputs, f32 accumulate.
// Workgroup tile 128x128, 8 waves, each wave 32x64 (2x4 WMMA tiles), K-step 32.
// A/W 128x32 slabs staged into LDS by the Tensor Data Mover, double-buffered
// so the next tile's DMA overlaps with WMMA on the current tile.
// mode 0: scatter qkv epilogue (Q,K as (BH,T,64) bf16; V transposed (BH,64,T)).
// mode 1: plain f32 row-major store to outf.
// ---------------------------------------------------------------------------
__global__ __launch_bounds__(256) void gemm_xwT(
    const __bf16* __restrict__ A, const __bf16* __restrict__ W,
    int M, int N, int K, int mode,
    __bf16* __restrict__ qb, __bf16* __restrict__ kb, __bf16* __restrict__ vtb,
    float* __restrict__ outf)
{
  extern __shared__ __bf16 smem[];   // [2 buffers][A: 128x32][W: 128x32]

  const int tid  = threadIdx.x;
  const int lane = tid & 31;
  const int half = lane >> 4;     // 0: lanes 0-15, 1: lanes 16-31
  const int l15  = lane & 15;
  const int wid  = tid >> 5;
  // Scalar (SGPR) wave id so TDM issue sits behind a *scalar* branch:
  // TDM ignores EXEC, so EXEC-predication would issue duplicate DMAs.
  const int swid = __builtin_amdgcn_readfirstlane(tid) >> 5;

  const int m0 = blockIdx.x * 128;
  const int n0 = blockIdx.y * 128;
  const int wm = (wid & 3) * 32;  // wave M sub-block
  const int wn = (wid >> 2) * 64; // wave N sub-block

  v8f acc[2][4];
  #pragma unroll
  for (int i = 0; i < 2; i++)
    #pragma unroll
    for (int j = 0; j < 4; j++)
      #pragma unroll
      for (int e = 0; e < 8; e++) acc[i][j][e] = 0.f;

  const int ksteps = K >> 5;

#if USE_TDM
  // ---- prologue: DMA the k=0 slabs into buffer 0
  if (swid == 0) {
    tdm_load_2d(A + (size_t)m0 * K, 0u,     32u, 128u, (unsigned)K);
    tdm_load_2d(W + (size_t)n0 * K, 8192u,  32u, 128u, (unsigned)K);
  }
  for (int ks = 0; ks < ksteps; ks++) {
    const int buf   = ks & 1;
    const int abase = buf * 8192;        // element offsets (16KB per buffer)
    const int wbase = abase + 4096;
    if (swid == 0) {
      __builtin_amdgcn_s_wait_tensorcnt(0);   // current buffer resident
    }
    __syncthreads();
    if (swid == 0 && ks + 1 < ksteps) {       // prefetch next (overlaps WMMA)
      const int kn = (ks + 1) << 5;
      const unsigned nb = (unsigned)((buf ^ 1) * 16384);
      tdm_load_2d(A + (size_t)m0 * K + kn, nb,         32u, 128u, (unsigned)K);
      tdm_load_2d(W + (size_t)n0 * K + kn, nb + 8192u, 32u, 128u, (unsigned)K);
    }
#else
  for (int ks = 0; ks < ksteps; ks++) {
    const int buf   = 0;
    const int abase = 0;
    const int wbase = 4096;
    {
      const int k0     = ks << 5;
      const int cprow  = tid >> 1;       // 0..127
      const int cphalf = (tid & 1) * 16; // element offset 0 or 16
      const uint4* ga = (const uint4*)(A + (size_t)(m0 + cprow) * K + k0 + cphalf);
      uint4 a0 = ga[0], a1 = ga[1];
      const uint4* gw = (const uint4*)(W + (size_t)(n0 + cprow) * K + k0 + cphalf);
      uint4 w0 = gw[0], w1 = gw[1];
      *(uint4*)&smem[abase + cprow * 32 + cphalf]     = a0;
      *(uint4*)&smem[abase + cprow * 32 + cphalf + 8] = a1;
      *(uint4*)&smem[wbase + cprow * 32 + cphalf]     = w0;
      *(uint4*)&smem[wbase + cprow * 32 + cphalf + 8] = w1;
    }
    __syncthreads();
#endif

    // ---- fragments + 8 WMMA on the current buffer
    v16bf afrag[2], bfrag[4];
    #pragma unroll
    for (int i = 0; i < 2; i++)
      afrag[i] = load_a_frag_lds(smem, abase, wm + i * 16 + l15, half);
    #pragma unroll
    for (int j = 0; j < 4; j++)
      bfrag[j] = load_b_frag_lds(smem, wbase, wn + j * 16 + l15, half);
    #pragma unroll
    for (int i = 0; i < 2; i++)
      #pragma unroll
      for (int j = 0; j < 4; j++)
        acc[i][j] = wmma_bf16(afrag[i], bfrag[j], acc[i][j]);
    __syncthreads();
  }

  // ---- epilogue: C layout lane=(col), vgpr r = row r (+8 for upper half)
  #pragma unroll
  for (int i = 0; i < 2; i++) {
    #pragma unroll
    for (int j = 0; j < 4; j++) {
      #pragma unroll
      for (int r = 0; r < 8; r++) {
        int m = m0 + wm + i * 16 + r + half * 8;
        int n = n0 + wn + j * 16 + l15;
        float v = acc[i][j][r];
        if (mode == 0) {
          int which = n >> 10;        // 0=q 1=k 2=v
          int c = n & 1023;
          int h = c >> 6, d = c & 63;
          int b = m >> 11, t = m & 2047;
          int bh = b * NHEADS + h;
          if (which == 0)      qb[((size_t)bh * SEQ + t) * HDIM + d]  = (__bf16)v;
          else if (which == 1) kb[((size_t)bh * SEQ + t) * HDIM + d]  = (__bf16)v;
          else                 vtb[((size_t)bh * HDIM + d) * SEQ + t] = (__bf16)v;
        } else {
          outf[(size_t)m * N + n] = v;
        }
      }
    }
  }
}

// ---------------------------------------------------------------------------
// Flash-style causal attention. Grid: (T/128, BH). 8 waves/WG, each wave owns
// a 16-row query tile; keys streamed in blocks of 32 with online softmax.
// Q,K: (BH, T, 64) bf16 ; Vt: (BH, 64, T) bf16 ; O: (B, T, H*64) bf16.
// ---------------------------------------------------------------------------
__global__ __launch_bounds__(256) void attn_fwd(
    const __bf16* __restrict__ Q, const __bf16* __restrict__ Km,
    const __bf16* __restrict__ Vt, __bf16* __restrict__ O)
{
  __shared__ __bf16 Ps[8][16][32];   // per-wave P slab (C-layout -> A-layout)

  const int tid  = threadIdx.x;
  const int lane = tid & 31;
  const int wid  = tid >> 5;
  const int half = lane >> 4;
  const int l15  = lane & 15;

  const int bh = blockIdx.y;
  const int q0 = blockIdx.x * 128 + wid * 16;

  const __bf16* Qb = Q  + (size_t)bh * SEQ * HDIM;
  const __bf16* Kb = Km + (size_t)bh * SEQ * HDIM;
  const __bf16* Vb = Vt + (size_t)bh * HDIM * SEQ;

  // Q A-fragments for d-chunks [0,32) and [32,64), kept in registers
  v16bf qf[2];
  #pragma unroll
  for (int i = 0; i < 2; i++) {
    const __bf16* p = Qb + (size_t)(q0 + l15) * HDIM + i * 32 + half * 16;
    v8bf lo = *(const v8bf*)p;
    v8bf hi = *(const v8bf*)(p + 8);
    #pragma unroll
    for (int e = 0; e < 8; e++) { qf[i][e] = lo[e]; qf[i][8 + e] = hi[e]; }
  }

  v8f   oacc[4];
  float mrow[8], lrow[8];
  #pragma unroll
  for (int dt = 0; dt < 4; dt++)
    #pragma unroll
    for (int e = 0; e < 8; e++) oacc[dt][e] = 0.f;
  #pragma unroll
  for (int r = 0; r < 8; r++) { mrow[r] = -1e30f; lrow[r] = 0.f; }

  const float scale = 0.125f;            // 1/sqrt(64)
  const int   nkb   = (q0 + 47) >> 5;    // causal: key blocks covering q0+15

  for (int kb2 = 0; kb2 < nkb; kb2++) {
    const int kbase = kb2 * 32;

    // ---- scores: two 16x16 tiles (keys kbase..+15, +16..+31), contract d=64
    v8f s[2];
    #pragma unroll
    for (int tt = 0; tt < 2; tt++) {
      #pragma unroll
      for (int e = 0; e < 8; e++) s[tt][e] = 0.f;
      #pragma unroll
      for (int ds = 0; ds < 2; ds++) {
        const __bf16* p = Kb + (size_t)(kbase + tt * 16 + l15) * HDIM
                             + ds * 32 + half * 16;
        v16bf bf = *(const v16bf*)p;
        s[tt] = wmma_bf16(qf[ds], bf, s[tt]);
      }
    }

    // ---- online softmax update (row stats align with per-lane C layout)
    #pragma unroll
    for (int r = 0; r < 8; r++) {
      int row = q0 + r + half * 8;
      float s0 = (kbase + l15)      <= row ? s[0][r] * scale : -1e30f;
      float s1 = (kbase + 16 + l15) <= row ? s[1][r] * scale : -1e30f;
      float bm = fmaxf(s0, s1);
      bm = fmaxf(bm, __shfl_xor(bm, 1, 32));
      bm = fmaxf(bm, __shfl_xor(bm, 2, 32));
      bm = fmaxf(bm, __shfl_xor(bm, 4, 32));
      bm = fmaxf(bm, __shfl_xor(bm, 8, 32));
      float mnew = fmaxf(mrow[r], bm);
      float corr = __expf(mrow[r] - mnew);
      float p0 = __expf(s0 - mnew);
      float p1 = __expf(s1 - mnew);
      float ps = p0 + p1;
      ps += __shfl_xor(ps, 1, 32);
      ps += __shfl_xor(ps, 2, 32);
      ps += __shfl_xor(ps, 4, 32);
      ps += __shfl_xor(ps, 8, 32);
      lrow[r] = lrow[r] * corr + ps;
      mrow[r] = mnew;
      #pragma unroll
      for (int dt = 0; dt < 4; dt++) oacc[dt][r] *= corr;
      int prow = r + half * 8;
      Ps[wid][prow][l15]      = (__bf16)p0;
      Ps[wid][prow][16 + l15] = (__bf16)p1;
    }
    asm volatile("s_wait_dscnt 0" ::: "memory");

    // ---- P: C-layout -> A-fragment via per-wave LDS slab
    v16bf pf;
    {
      int base = half * 16;
      v8bf lo = *(const v8bf*)&Ps[wid][l15][base];
      v8bf hi = *(const v8bf*)&Ps[wid][l15][base + 8];
      #pragma unroll
      for (int e = 0; e < 8; e++) { pf[e] = lo[e]; pf[8 + e] = hi[e]; }
    }

    // ---- O += P @ V  (V transposed: contraction-contiguous B-fragments)
    #pragma unroll
    for (int dt = 0; dt < 4; dt++) {
      const __bf16* p = Vb + (size_t)(dt * 16 + l15) * SEQ + kbase + half * 16;
      v16bf bf = *(const v16bf*)p;
      oacc[dt] = wmma_bf16(pf, bf, oacc[dt]);
    }
  }

  // ---- normalize and store O as (B, T, H*64) bf16
  const int b = bh >> 4, h = bh & 15;
  #pragma unroll
  for (int r = 0; r < 8; r++) {
    float inv = 1.0f / lrow[r];
    int t = q0 + r + half * 8;
    size_t base = ((size_t)(b * SEQ + t)) * DMODEL + h * HDIM;
    #pragma unroll
    for (int dt = 0; dt < 4; dt++)
      O[base + dt * 16 + l15] = (__bf16)(oacc[dt][r] * inv);
  }
}

// ---------------------------------------------------------------------------
// Launch
// ---------------------------------------------------------------------------
extern "C" void kernel_launch(void* const* d_in, const int* in_sizes, int n_in,
                              void* d_out, int out_size, void* d_ws, size_t ws_size,
                              hipStream_t stream) {
  const float* x     = (const float*)d_in[0];  // (4,2048,1024)
  const float* w_qkv = (const float*)d_in[1];  // (3072,1024)
  const float* w_out = (const float*)d_in[2];  // (1024,1024)
  float* out = (float*)d_out;                  // (4,2048,1024)

  char* ws = (char*)d_ws;
  size_t off = 0;
  auto carve = [&](size_t bytes) -> char* {
    char* p = ws + off;
    off += (bytes + 255) & ~(size_t)255;
    return p;
  };
  const size_t NX    = (size_t)MTOK * DMODEL;      // 8388608
  const size_t NWQKV = (size_t)3 * DMODEL * DMODEL;
  const size_t NWOUT = (size_t)DMODEL * DMODEL;
  const size_t NQKV  = (size_t)BH * SEQ * HDIM;    // 8388608 per tensor

  __bf16* xb   = (__bf16*)carve(NX * 2);
  __bf16* wqb  = (__bf16*)carve(NWQKV * 2);
  __bf16* wob  = (__bf16*)carve(NWOUT * 2);
  __bf16* qb   = (__bf16*)carve(NQKV * 2);
  __bf16* kb   = (__bf16*)carve(NQKV * 2);
  __bf16* vtb  = (__bf16*)carve(NQKV * 2);
  __bf16* ob   = (__bf16*)carve(NX * 2);

  const size_t gemm_lds = 2 * 16384;   // double-buffered A/W slabs

  // 1) precision conversion
  cvt_f32_bf16<<<(int)(NX / 4 / 256),    256, 0, stream>>>(x,     xb,  (int)NX);
  cvt_f32_bf16<<<(int)(NWQKV / 4 / 256), 256, 0, stream>>>(w_qkv, wqb, (int)NWQKV);
  cvt_f32_bf16<<<(int)(NWOUT / 4 / 256), 256, 0, stream>>>(w_out, wob, (int)NWOUT);

  // 2) qkv = x @ w_qkv^T, scattered to Q, K, V^T (bf16)
  gemm_xwT<<<dim3(MTOK / 128, (3 * DMODEL) / 128), 256, gemm_lds, stream>>>(
      xb, wqb, MTOK, 3 * DMODEL, DMODEL, /*mode=*/0, qb, kb, vtb, nullptr);

  // 3) causal flash attention
  attn_fwd<<<dim3(SEQ / 128, BH), 256, 0, stream>>>(qb, kb, vtb, ob);

  // 4) out = O @ w_out^T (f32 store)
  gemm_xwT<<<dim3(MTOK / 128, DMODEL / 128), 256, gemm_lds, stream>>>(
      ob, wob, MTOK, DMODEL, DMODEL, /*mode=*/1, nullptr, nullptr, nullptr, out);
}
